// FoodBankGNN_14242111553545
// MI455X (gfx1250) — matrix operand order repaired and checked by reading, not compiled, not run on hardware
//
#include <hip/hip_runtime.h>
#include <hip/hip_bf16.h>

#define NN 50000
#define NE 1600000
#define IND 16

typedef __attribute__((ext_vector_type(16))) _Float16 v16h;
typedef __attribute__((ext_vector_type(8)))  _Float16 v8h;
typedef __attribute__((ext_vector_type(8)))  float    v8f;

union AFrag { v16h v; v8h g[2]; _Float16 e[16]; };
union CFrag { v8f  v; float e[8]; };

__device__ __forceinline__ float lrelu02(float x){ return x > 0.f ? x : 0.2f * x; }
__device__ __forceinline__ unsigned f2ord(float f){
    unsigned u = __float_as_uint(f);
    return (u & 0x80000000u) ? ~u : (u | 0x80000000u);
}
__device__ __forceinline__ float ord2f(unsigned u){
    return __uint_as_float((u & 0x80000000u) ? (u & 0x7fffffffu) : ~u);
}

// ---------------- f32 -> f16 transposed convert: dst[n][k] = src[k][n] ----------------
__global__ void k_cvt_tr(const float* __restrict__ src, _Float16* __restrict__ dst,
                         int K, int Nc){
    int i = blockIdx.x * blockDim.x + threadIdx.x;
    if (i >= K * Nc) return;
    int k = i / Nc, n = i % Nc;
    dst[n * K + k] = (_Float16)src[i];
}

// ---------------- GAT layer 1: h1 = x@W1 (K=16, VALU), fused alpha dots ----------------
__global__ void k_layer1(const float* __restrict__ x, const float* __restrict__ W1,
                         const float* __restrict__ as1, const float* __restrict__ ad1,
                         float* __restrict__ h1, float* __restrict__ asv, float* __restrict__ adv){
    int n = blockIdx.x * blockDim.x + threadIdx.x;
    if (n >= NN) return;
    float xv[IND];
#pragma unroll
    for (int i = 0; i < IND; ++i) xv[i] = x[n*IND + i];
    float asum0 = 0.f, asum1 = 0.f, adsum0 = 0.f, adsum1 = 0.f;
    for (int j = 0; j < 128; ++j){
        float acc = 0.f;
#pragma unroll
        for (int i = 0; i < IND; ++i) acc += xv[i] * W1[i*128 + j];
        h1[n*128 + j] = acc;
        if (j < 64){ asum0 += acc * as1[j]; adsum0 += acc * ad1[j]; }
        else       { asum1 += acc * as1[j]; adsum1 += acc * ad1[j]; }
    }
    asv[n*2+0] = asum0; asv[n*2+1] = asum1;
    adv[n*2+0] = adsum0; adv[n*2+1] = adsum1;
}

// ---------------- segment softmax layer 1 ----------------
__global__ void k_selfmax1(const float* __restrict__ asv, const float* __restrict__ adv,
                           unsigned* __restrict__ m1){
    int n = blockIdx.x * blockDim.x + threadIdx.x;
    if (n >= NN) return;
    m1[n*2+0] = f2ord(lrelu02(asv[n*2+0] + adv[n*2+0]));
    m1[n*2+1] = f2ord(lrelu02(asv[n*2+1] + adv[n*2+1]));
}
__global__ void k_edgemax1(const int* __restrict__ ei, const float* __restrict__ asv,
                           const float* __restrict__ adv, unsigned* __restrict__ m1){
    int e = blockIdx.x * blockDim.x + threadIdx.x;
    if (e >= NE) return;
    int s = ei[e], d = ei[NE + e];
    atomicMax(&m1[d*2+0], f2ord(lrelu02(asv[s*2+0] + adv[d*2+0])));
    atomicMax(&m1[d*2+1], f2ord(lrelu02(asv[s*2+1] + adv[d*2+1])));
}
__global__ void k_selfagg1(const float* __restrict__ asv, const float* __restrict__ adv,
                           const unsigned* __restrict__ m1, const float* __restrict__ h1,
                           float* __restrict__ den, float* __restrict__ agg){
    int n = blockIdx.x * blockDim.x + threadIdx.x;
    if (n >= NN) return;
    float w0 = __expf(lrelu02(asv[n*2+0] + adv[n*2+0]) - ord2f(m1[n*2+0]));
    float w1 = __expf(lrelu02(asv[n*2+1] + adv[n*2+1]) - ord2f(m1[n*2+1]));
    den[n*2+0] = w0; den[n*2+1] = w1;
    for (int c = 0; c < 128; ++c)
        agg[n*128 + c] = h1[n*128 + c] * ((c < 64) ? w0 : w1);
}
__global__ void k_edgeagg1(const int* __restrict__ ei, const float* __restrict__ asv,
                           const float* __restrict__ adv, const unsigned* __restrict__ m1,
                           const float* __restrict__ h1, float* __restrict__ den,
                           float* __restrict__ agg){
    long long t = (long long)blockIdx.x * blockDim.x + threadIdx.x;
    if (t >= (long long)NE * 128) return;
    int e = (int)(t >> 7), c = (int)(t & 127), hd = c >> 6;
    int s = ei[e], d = ei[NE + e];
    float w = __expf(lrelu02(asv[s*2+hd] + adv[d*2+hd]) - ord2f(m1[d*2+hd]));
    if ((c & 63) == 0) atomicAdd(&den[d*2+hd], w);
    atomicAdd(&agg[d*128 + c], w * h1[s*128 + c]);
}
__global__ void k_fin1(const float* __restrict__ agg, const float* __restrict__ den,
                       const float* __restrict__ b1, _Float16* __restrict__ h1h){
    int t = blockIdx.x * blockDim.x + threadIdx.x;
    if (t >= NN * 128) return;
    int n = t >> 7, c = t & 127;
    float v = agg[t] / den[n*2 + (c >> 6)] + b1[c];
    h1h[t] = (_Float16)(v > 0.f ? v : 0.f);
}

// ---------------- layer2 GEMM (WMMA): h2pre = h1h[50000x128] @ W2 (B from W2t[64x128]) ----------------
__global__ __launch_bounds__(32) void k_gemm2(const _Float16* __restrict__ h1h,
                                              const _Float16* __restrict__ W2t,
                                              float* __restrict__ h2pre){
    int lane = threadIdx.x, row = lane & 15, hv = lane >> 4, col = lane & 15;
    int rowbase = blockIdx.x * 16;
    const _Float16* arow = h1h + (size_t)(rowbase + row) * 128;
    CFrag acc[4];
#pragma unroll
    for (int n0 = 0; n0 < 4; ++n0)
#pragma unroll
        for (int r = 0; r < 8; ++r) acc[n0].e[r] = 0.f;
#pragma unroll
    for (int kc = 0; kc < 4; ++kc){
        AFrag a;
        a.g[0] = *(const v8h*)(arow + kc*32 + hv*8);
        a.g[1] = *(const v8h*)(arow + kc*32 + 16 + hv*8);
#pragma unroll
        for (int n0 = 0; n0 < 4; ++n0){
            AFrag b;
            const _Float16* bp = W2t + (size_t)(n0*16 + col) * 128 + kc*32 + hv*8;
            b.g[0] = *(const v8h*)(bp);
            b.g[1] = *(const v8h*)(bp + 16);
            acc[n0].v = __builtin_amdgcn_wmma_f32_16x16x32_f16(
                false, a.v, false, b.v, (short)0, acc[n0].v, false, false);
        }
    }
#pragma unroll
    for (int n0 = 0; n0 < 4; ++n0)
#pragma unroll
        for (int r = 0; r < 8; ++r)
            h2pre[(size_t)(rowbase + hv*8 + r) * 64 + n0*16 + col] = acc[n0].e[r];
}

// ---------------- layer2 attention (1 head, 64ch) ----------------
__global__ void k_alpha2(const float* __restrict__ h2, const float* __restrict__ as2,
                         const float* __restrict__ ad2, float* __restrict__ asv,
                         float* __restrict__ adv){
    int n = blockIdx.x * blockDim.x + threadIdx.x;
    if (n >= NN) return;
    float a = 0.f, b = 0.f;
    for (int c = 0; c < 64; ++c){ float h = h2[n*64 + c]; a += h * as2[c]; b += h * ad2[c]; }
    asv[n] = a; adv[n] = b;
}
__global__ void k_selfmax2(const float* __restrict__ asv, const float* __restrict__ adv,
                           unsigned* __restrict__ m2){
    int n = blockIdx.x * blockDim.x + threadIdx.x;
    if (n >= NN) return;
    m2[n] = f2ord(lrelu02(asv[n] + adv[n]));
}
__global__ void k_edgemax2(const int* __restrict__ ei, const float* __restrict__ asv,
                           const float* __restrict__ adv, unsigned* __restrict__ m2){
    int e = blockIdx.x * blockDim.x + threadIdx.x;
    if (e >= NE) return;
    int s = ei[e], d = ei[NE + e];
    atomicMax(&m2[d], f2ord(lrelu02(asv[s] + adv[d])));
}
__global__ void k_selfagg2(const float* __restrict__ asv, const float* __restrict__ adv,
                           const unsigned* __restrict__ m2, const float* __restrict__ h2,
                           float* __restrict__ den, float* __restrict__ agg){
    int n = blockIdx.x * blockDim.x + threadIdx.x;
    if (n >= NN) return;
    float w = __expf(lrelu02(asv[n] + adv[n]) - ord2f(m2[n]));
    den[n] = w;
    for (int c = 0; c < 64; ++c) agg[n*64 + c] = h2[n*64 + c] * w;
}
__global__ void k_edgeagg2(const int* __restrict__ ei, const float* __restrict__ asv,
                           const float* __restrict__ adv, const unsigned* __restrict__ m2,
                           const float* __restrict__ h2, float* __restrict__ den,
                           float* __restrict__ agg){
    long long t = (long long)blockIdx.x * blockDim.x + threadIdx.x;
    if (t >= (long long)NE * 64) return;
    int e = (int)(t >> 6), c = (int)(t & 63);
    int s = ei[e], d = ei[NE + e];
    float w = __expf(lrelu02(asv[s] + adv[d]) - ord2f(m2[d]));
    if (c == 0) atomicAdd(&den[d], w);
    atomicAdd(&agg[d*64 + c], w * h2[s*64 + c]);
}
__global__ void k_fin2(const float* __restrict__ agg, const float* __restrict__ den,
                       const float* __restrict__ b2, _Float16* __restrict__ h2h){
    int t = blockIdx.x * blockDim.x + threadIdx.x;
    if (t >= NN * 64) return;
    int n = t >> 6, c = t & 63;
    float v = agg[t] / den[n] + b2[c];
    h2h[t] = (_Float16)(v > 0.f ? v : 0.f);
}

// ---------------- edge MLP (WMMA): [h2[s]|h2[d]] -> 64 -> 64 -> 64 -> 1 ----------------
// 4 waves/block, 25 tiles of 16 edges per wave, weights staged once in LDS (transposed).
#define MLP_TPW 25
__global__ __launch_bounds__(128) void k_mlp(const _Float16* __restrict__ h2h,
                                             const int* __restrict__ ei,
                                             const _Float16* __restrict__ mW1t, const float* __restrict__ mb1,
                                             const _Float16* __restrict__ mW2t, const float* __restrict__ mb2,
                                             const _Float16* __restrict__ mW3t, const float* __restrict__ mb3,
                                             const float* __restrict__ mW4, const float* __restrict__ mb4,
                                             float* __restrict__ out){
    __shared__ __attribute__((aligned(16))) _Float16 sW1[64*128]; // transposed [n][k]
    __shared__ __attribute__((aligned(16))) _Float16 sW2[64*64];
    __shared__ __attribute__((aligned(16))) _Float16 sW3[64*64];
    __shared__ __attribute__((aligned(16))) float    sW4[64];
    __shared__ __attribute__((aligned(16))) _Float16 sZa[4][16*64]; // per-wave staging

    int tid = threadIdx.x;
    for (int i = tid; i < 64*128/8; i += 128) ((uint4*)sW1)[i] = ((const uint4*)mW1t)[i];
    for (int i = tid; i < 64*64/8;  i += 128){
        ((uint4*)sW2)[i] = ((const uint4*)mW2t)[i];
        ((uint4*)sW3)[i] = ((const uint4*)mW3t)[i];
    }
    if (tid < 16) ((uint4*)sW4)[tid] = ((const uint4*)mW4)[tid];
    __syncthreads();

    int wid = tid >> 5, lane = tid & 31;
    int row = lane & 15, hv = lane >> 4, col = lane & 15;
    _Float16* sZ = sZa[wid];
    float bias4 = mb4[0];

    for (int t = 0; t < MLP_TPW; ++t){
        int ebase = ((blockIdx.x * 4 + wid) * MLP_TPW + t) * 16;
        int er = ebase + row;
        int s = ei[er], d = ei[NE + er];
        const _Float16* hs = h2h + (size_t)s * 64;
        const _Float16* hd = h2h + (size_t)d * 64;

        // ---- layer 1: 16x128 @ 128x64 ----
        CFrag acc[4];
#pragma unroll
        for (int n0 = 0; n0 < 4; ++n0)
#pragma unroll
            for (int r = 0; r < 8; ++r) acc[n0].e[r] = 0.f;
#pragma unroll
        for (int kc = 0; kc < 4; ++kc){
            AFrag a;
            int k0 = kc*32 + hv*8;
            int k1 = kc*32 + 16 + hv*8;
            a.g[0] = (k0 < 64) ? *(const v8h*)(hs + k0) : *(const v8h*)(hd + (k0 - 64));
            a.g[1] = (k1 < 64) ? *(const v8h*)(hs + k1) : *(const v8h*)(hd + (k1 - 64));
#pragma unroll
            for (int n0 = 0; n0 < 4; ++n0){
                AFrag b;
                const _Float16* bp = sW1 + (n0*16 + col)*128 + kc*32 + hv*8;
                b.g[0] = *(const v8h*)(bp);
                b.g[1] = *(const v8h*)(bp + 16);
                acc[n0].v = __builtin_amdgcn_wmma_f32_16x16x32_f16(
                    false, a.v, false, b.v, (short)0, acc[n0].v, false, false);
            }
        }
#pragma unroll
        for (int n0 = 0; n0 < 4; ++n0){
            float bb = mb1[n0*16 + col];
#pragma unroll
            for (int r = 0; r < 8; ++r){
                float v = acc[n0].e[r] + bb;
                sZ[(hv*8 + r) * 64 + n0*16 + col] = (_Float16)(v > 0.f ? v : 0.f);
            }
        }
        __syncthreads();

        // ---- layers 2 and 3: 16x64 @ 64x64 ----
#pragma unroll
        for (int L = 0; L < 2; ++L){
            const _Float16* sw = (L == 0) ? sW2 : sW3;
            const float* mb = (L == 0) ? mb2 : mb3;
            CFrag acc2[4];
#pragma unroll
            for (int n0 = 0; n0 < 4; ++n0)
#pragma unroll
                for (int r = 0; r < 8; ++r) acc2[n0].e[r] = 0.f;
#pragma unroll
            for (int kc = 0; kc < 2; ++kc){
                AFrag a;
                a.g[0] = *(const v8h*)(sZ + row*64 + kc*32 + hv*8);
                a.g[1] = *(const v8h*)(sZ + row*64 + kc*32 + 16 + hv*8);
#pragma unroll
                for (int n0 = 0; n0 < 4; ++n0){
                    AFrag b;
                    const _Float16* bp = sw + (n0*16 + col)*64 + kc*32 + hv*8;
                    b.g[0] = *(const v8h*)(bp);
                    b.g[1] = *(const v8h*)(bp + 16);
                    acc2[n0].v = __builtin_amdgcn_wmma_f32_16x16x32_f16(
                        false, a.v, false, b.v, (short)0, acc2[n0].v, false, false);
                }
            }
            __syncthreads();
#pragma unroll
            for (int n0 = 0; n0 < 4; ++n0){
                float bb = mb[n0*16 + col];
#pragma unroll
                for (int r = 0; r < 8; ++r){
                    float v = acc2[n0].e[r] + bb;
                    sZ[(hv*8 + r) * 64 + n0*16 + col] = (_Float16)(v > 0.f ? v : 0.f);
                }
            }
            __syncthreads();
        }

        // ---- layer 4: 16x64 @ 64x1 ----
        if (lane < 16){
            float o = bias4;
#pragma unroll
            for (int c = 0; c < 64; ++c) o += (float)sZ[lane*64 + c] * sW4[c];
            out[ebase + lane] = o;
        }
        __syncthreads();
    }
}

extern "C" void kernel_launch(void* const* d_in, const int* in_sizes, int n_in,
                              void* d_out, int out_size, void* d_ws, size_t ws_size,
                              hipStream_t stream) {
    (void)in_sizes; (void)n_in; (void)out_size; (void)ws_size;
    const float* x   = (const float*)d_in[0];
    const int*   ei  = (const int*)d_in[1];
    const float* W1  = (const float*)d_in[2];
    const float* as1 = (const float*)d_in[3];
    const float* ad1 = (const float*)d_in[4];
    const float* b1  = (const float*)d_in[5];
    const float* W2  = (const float*)d_in[6];
    const float* as2 = (const float*)d_in[7];
    const float* ad2 = (const float*)d_in[8];
    const float* b2  = (const float*)d_in[9];
    const float* mW1 = (const float*)d_in[10];
    const float* mb1 = (const float*)d_in[11];
    const float* mW2 = (const float*)d_in[12];
    const float* mb2 = (const float*)d_in[13];
    const float* mW3 = (const float*)d_in[14];
    const float* mb3 = (const float*)d_in[15];
    const float* mW4 = (const float*)d_in[16];
    const float* mb4 = (const float*)d_in[17];
    float* out = (float*)d_out;

    char* base = (char*)d_ws; size_t off = 0;
    auto alloc = [&](size_t bytes) -> void* {
        void* p = base + off; off += (bytes + 255) & ~(size_t)255; return p;
    };
    float*     h1    = (float*)alloc((size_t)NN * 128 * 4);
    float*     asv1  = (float*)alloc((size_t)NN * 2 * 4);
    float*     adv1  = (float*)alloc((size_t)NN * 2 * 4);
    unsigned*  m1    = (unsigned*)alloc((size_t)NN * 2 * 4);
    float*     den1  = (float*)alloc((size_t)NN * 2 * 4);
    float*     agg1  = (float*)alloc((size_t)NN * 128 * 4);
    _Float16*  h1h   = (_Float16*)alloc((size_t)NN * 128 * 2);
    _Float16*  W2t   = (_Float16*)alloc((size_t)64 * 128 * 2);
    float*     h2pre = (float*)alloc((size_t)NN * 64 * 4);
    float*     asv2  = (float*)alloc((size_t)NN * 4);
    float*     adv2  = (float*)alloc((size_t)NN * 4);
    unsigned*  m2    = (unsigned*)alloc((size_t)NN * 4);
    float*     den2  = (float*)alloc((size_t)NN * 4);
    float*     agg2  = (float*)alloc((size_t)NN * 64 * 4);
    _Float16*  h2h   = (_Float16*)alloc((size_t)NN * 64 * 2);
    _Float16*  mW1t  = (_Float16*)alloc((size_t)64 * 128 * 2);
    _Float16*  mW2t  = (_Float16*)alloc((size_t)64 * 64 * 2);
    _Float16*  mW3t  = (_Float16*)alloc((size_t)64 * 64 * 2);

    // transposed f16 weight conversions (dst[n][k] = src[k][n])
    k_cvt_tr<<<(128*64 + 255)/256, 256, 0, stream>>>(W2,  W2t,  128, 64);
    k_cvt_tr<<<(128*64 + 255)/256, 256, 0, stream>>>(mW1, mW1t, 128, 64);
    k_cvt_tr<<<(64*64 + 255)/256, 256, 0, stream>>>(mW2, mW2t, 64, 64);
    k_cvt_tr<<<(64*64 + 255)/256, 256, 0, stream>>>(mW3, mW3t, 64, 64);

    // GAT layer 1
    k_layer1<<<(NN + 127)/128, 128, 0, stream>>>(x, W1, as1, ad1, h1, asv1, adv1);
    k_selfmax1<<<(NN + 255)/256, 256, 0, stream>>>(asv1, adv1, m1);
    k_edgemax1<<<NE/256, 256, 0, stream>>>(ei, asv1, adv1, m1);
    k_selfagg1<<<(NN + 255)/256, 256, 0, stream>>>(asv1, adv1, m1, h1, den1, agg1);
    k_edgeagg1<<<(NE*128LL)/256, 256, 0, stream>>>(ei, asv1, adv1, m1, h1, den1, agg1);
    k_fin1<<<(NN*128)/256, 256, 0, stream>>>(agg1, den1, b1, h1h);

    // GAT layer 2
    k_gemm2<<<NN/16, 32, 0, stream>>>(h1h, W2t, h2pre);
    k_alpha2<<<(NN + 127)/128, 128, 0, stream>>>(h2pre, as2, ad2, asv2, adv2);
    k_selfmax2<<<(NN + 255)/256, 256, 0, stream>>>(asv2, adv2, m2);
    k_edgemax2<<<NE/256, 256, 0, stream>>>(ei, asv2, adv2, m2);
    k_selfagg2<<<(NN + 255)/256, 256, 0, stream>>>(asv2, adv2, m2, h2pre, den2, agg2);
    k_edgeagg2<<<(NE*64LL)/256, 256, 0, stream>>>(ei, asv2, adv2, m2, h2pre, den2, agg2);
    k_fin2<<<(NN*64)/256, 256, 0, stream>>>(agg2, den2, b2, h2h);

    // Edge MLP: 100000 tiles = 1000 blocks * 4 waves * 25 tiles
    k_mlp<<<(NE/16)/(4*MLP_TPW), 128, 0, stream>>>(h2h, ei, mW1t, mb1, mW2t, mb2,
                                                   mW3t, mb3, mW4, mb4, out);
}
